// DotProductAttention_50010599194781
// MI455X (gfx1250) — matrix operand order repaired and checked by reading, not compiled, not run on hardware
//
#include <hip/hip_runtime.h>
#include <hip/hip_bf16.h>
#include <cstdint>

// ---------------------------------------------------------------------------
// CDNA5 (gfx1250) flash-attention with post-softmax multiplicative term.
// GEMMs on v_wmma_f32_16x16x32_bf16; K/V/kv tiles staged in LDS per block via
// async global->LDS loads (ASYNCcnt); row reductions via DPP row_ror.
// ---------------------------------------------------------------------------

typedef __attribute__((ext_vector_type(16))) __bf16 v16bf;
typedef __attribute__((ext_vector_type(8)))  float  v8f;
typedef __attribute__((ext_vector_type(4)))  int    v4i;

#define WMMA_BF16(a, b, c) \
  __builtin_amdgcn_wmma_f32_16x16x32_bf16(false, (a), false, (b), (short)0, (c), false, false)

constexpr int B_  = 2;
constexpr int H_  = 16;
constexpr int BH_ = 32;
constexpr int L_  = 2048;
constexpr int D_  = 128;
constexpr float NEGV  = -1e6f;
constexpr float SCALE = 0.08838834764831845f;   // 1/sqrt(128)

union Frag { v16bf v; uint4 u[2]; };

// 16 contiguous bf16: B-fragment (lane = column n, elems k = half*16 + 0..15).
__device__ inline v16bf load_b32frag(const __bf16* p) {
  Frag f;
  f.u[0] = *(const uint4*)(p);
  f.u[1] = *(const uint4*)(p + 8);
  return f.v;
}

// A-fragment (16x32, 16-bit): lane (half, lr) -> row lr,
// elems 0..15 = K = d0 + half*16 + 0..15.
__device__ inline v16bf load_afrag(const __bf16* row, int d0, int half) {
  Frag f;
  f.u[0] = *(const uint4*)(row + d0 + half * 16);
  f.u[1] = *(const uint4*)(row + d0 + half * 16 + 8);
  return f.v;
}

// DPP row rotate (within 16-lane row): ctrl = 0x120 | N.
template <int CTRL>
__device__ inline float dpp_ror(float x) {
  int r = __builtin_amdgcn_update_dpp(0, __builtin_bit_cast(int, x), CTRL, 0xf, 0xf, true);
  return __builtin_bit_cast(float, r);
}
__device__ inline float rowred_max(float x) {
  x = fmaxf(x, dpp_ror<0x128>(x));
  x = fmaxf(x, dpp_ror<0x124>(x));
  x = fmaxf(x, dpp_ror<0x122>(x));
  x = fmaxf(x, dpp_ror<0x121>(x));
  return x;
}
__device__ inline float rowred_add(float x) {
  x += dpp_ror<0x128>(x);
  x += dpp_ror<0x124>(x);
  x += dpp_ror<0x122>(x);
  x += dpp_ror<0x121>(x);
  return x;
}

// ---- async global -> LDS copy (16B granules), with safe fallback ----------
#if defined(__AMDGCN__) && __has_builtin(__builtin_amdgcn_global_load_async_to_lds_b128)
#define HAVE_ASYNC_LDS 1
#endif

__device__ inline void cp16_g2l(const __bf16* g, __bf16* l) {
#ifdef HAVE_ASYNC_LDS
  __builtin_amdgcn_global_load_async_to_lds_b128(
      (__attribute__((address_space(1))) v4i*)(g),
      (__attribute__((address_space(3))) v4i*)(l), 0, 0);
#else
  *(uint4*)l = *(const uint4*)g;
#endif
}
__device__ inline void async_copies_done() {
#ifdef HAVE_ASYNC_LDS
  asm volatile("s_wait_asynccnt 0" ::: "memory");
#endif
}

// ---------------------------------------------------------------------------
// Pass 1: fp32 -> bf16; V transposed to [bh][d][k] for contiguous B-fragments.
// ---------------------------------------------------------------------------
__global__ void convert_kernel(const float* __restrict__ Q, const float* __restrict__ K,
                               const float* __restrict__ V, const float* __restrict__ qm,
                               const float* __restrict__ kvm,
                               __bf16* __restrict__ Qb, __bf16* __restrict__ Kb,
                               __bf16* __restrict__ Vt, __bf16* __restrict__ qmb,
                               __bf16* __restrict__ kvmb) {
  const long long NQ = (long long)BH_ * L_ * D_;
  const long long NM = (long long)B_ * L_ * D_;
  const long long total = 3 * NQ + 2 * NM;
  for (long long i = (long long)blockIdx.x * blockDim.x + threadIdx.x; i < total;
       i += (long long)gridDim.x * blockDim.x) {
    if (i < NQ) {
      Qb[i] = (__bf16)Q[i];
    } else if (i < 2 * NQ) {
      Kb[i - NQ] = (__bf16)K[i - NQ];
    } else if (i < 3 * NQ) {
      long long j = i - 2 * NQ;
      int bh = (int)(j / (L_ * D_));
      int r  = (int)(j % (L_ * D_));
      int k  = r / D_;
      int d  = r % D_;
      Vt[((long long)bh * D_ + d) * L_ + k] = (__bf16)V[j];
    } else if (i < 3 * NQ + NM) {
      qmb[i - 3 * NQ] = (__bf16)qm[i - 3 * NQ];
    } else {
      kvmb[i - 3 * NQ - NM] = (__bf16)kvm[i - 3 * NQ - NM];
    }
  }
}

// ---------------------------------------------------------------------------
// Pass 2a: per-16x16-tile sum of squares of M = qm . kvm^T (WMMA, K=128).
// ---------------------------------------------------------------------------
__global__ void fro_partial_kernel(const __bf16* __restrict__ qmb,
                                   const __bf16* __restrict__ kvmb,
                                   float* __restrict__ partial) {
  const int wid  = (int)((blockIdx.x * blockDim.x + threadIdx.x) >> 5);
  const int lane = threadIdx.x & 31;
  const int half = lane >> 4;
  const int lr   = lane & 15;
  const int b    = wid / (128 * 128);
  const int rem  = wid % (128 * 128);
  const int qt   = rem / 128;
  const int kt   = rem % 128;

  const __bf16* qrow = qmb  + ((long long)b * L_ + qt * 16 + lr) * D_;
  const __bf16* krow = kvmb + ((long long)b * L_ + kt * 16 + lr) * D_;

  v8f acc = {};
#pragma unroll
  for (int i = 0; i < 4; ++i) {
    v16bf a  = load_afrag(qrow, i * 32, half);
    v16bf bb = load_b32frag(krow + i * 32 + half * 16);
    acc = WMMA_BF16(a, bb, acc);
  }
  float s = 0.f;
#pragma unroll
  for (int j = 0; j < 8; ++j) s += acc[j] * acc[j];
  s = rowred_add(s);
  int sw = __builtin_amdgcn_ds_swizzle(__builtin_bit_cast(int, s), 0x401f);  // SWAPX16
  s += __builtin_bit_cast(float, sw);
  if (lane == 0) partial[wid] = s;
}

// Pass 2b: deterministic tree reduction of 128*128 tile partials per batch.
__global__ void fro_reduce_kernel(const float* __restrict__ partial, float* __restrict__ fro) {
  __shared__ float sm[256];
  const int b = blockIdx.x;
  float s = 0.f;
  for (int i = threadIdx.x; i < 128 * 128; i += 256) s += partial[b * 128 * 128 + i];
  sm[threadIdx.x] = s;
  __syncthreads();
  for (int st = 128; st >= 1; st >>= 1) {
    if ((int)threadIdx.x < st) sm[threadIdx.x] += sm[threadIdx.x + st];
    __syncthreads();
  }
  if (threadIdx.x == 0) fro[b] = sm[0];
}

// ---------------------------------------------------------------------------
// Pass 3: attention. Block = 8 waves x 16 queries, keys in 64-chunks staged
// into LDS once per block (async), shared by all 8 waves.
// ---------------------------------------------------------------------------
__global__ __launch_bounds__(256)
void attn_kernel(const __bf16* __restrict__ Qb, const __bf16* __restrict__ Kb,
                 const __bf16* __restrict__ Vt, const __bf16* __restrict__ qmb,
                 const __bf16* __restrict__ kvmb, const int* __restrict__ valid,
                 const float* __restrict__ fro, float* __restrict__ out) {
  __shared__ __bf16 sK[64 * 128];     // 16KB  [k_local][d]
  __shared__ __bf16 sKV[64 * 128];    // 16KB  [k_local][d]
  __shared__ __bf16 sV[128 * 64];     // 16KB  [d][k_local]
  __shared__ __bf16 sP[8 * 16 * 64];  // 16KB  per-wave P staging

  const int tid  = threadIdx.x;
  const int bh   = blockIdx.x >> 4;
  const int qblk = blockIdx.x & 15;
  const int wave = tid >> 5;
  const int lane = tid & 31;
  const int half = lane >> 4;
  const int lr   = lane & 15;
  const int b    = bh >> 4;              // bh / num_heads
  const int qbase = qblk * 128 + wave * 16;

  const int   vl   = valid[bh];
  const float rfro = 1.0f / (sqrtf(fro[b]) + 1e-5f);

  const __bf16* qrow  = Qb  + ((long long)bh * L_ + qbase + lr) * D_;
  const __bf16* qmrow = qmb + ((long long)b  * L_ + qbase + lr) * D_;
  v16bf aq[4], aqm[4];
#pragma unroll
  for (int i = 0; i < 4; ++i) {
    aq[i]  = load_afrag(qrow,  i * 32, half);
    aqm[i] = load_afrag(qmrow, i * 32, half);
  }

  const __bf16* Kbh = Kb   + (long long)bh * L_ * D_;
  const __bf16* kvb = kvmb + (long long)b  * L_ * D_;
  const __bf16* Vth = Vt   + (long long)bh * D_ * L_;

  float mrow[8], lrow[8];
#pragma unroll
  for (int j = 0; j < 8; ++j) { mrow[j] = -INFINITY; lrow[j] = 0.f; }

  // ---- Sweep 1: masked softmax stats; K chunk staged in LDS ------------
  for (int k0 = 0; k0 < L_; k0 += 64) {
    __syncthreads();                         // previous chunk fully consumed
#pragma unroll
    for (int q = 0; q < 4; ++q) {            // 64x128 bf16 contiguous = 16KB
      int e = (tid * 4 + q) * 8;
      cp16_g2l(Kbh + (long long)k0 * D_ + e, sK + e);
    }
    async_copies_done();
    __syncthreads();

    v8f S[4] = {};
#pragma unroll
    for (int t = 0; t < 4; ++t) {
      const __bf16* krow = sK + (t * 16 + lr) * D_;
#pragma unroll
      for (int i = 0; i < 4; ++i)
        S[t] = WMMA_BF16(aq[i], load_b32frag(krow + i * 32 + half * 16), S[t]);
    }
    bool c[4];
#pragma unroll
    for (int t = 0; t < 4; ++t) c[t] = (k0 + t * 16 + lr) < vl;
#pragma unroll
    for (int j = 0; j < 8; ++j) {
      float s[4];
#pragma unroll
      for (int t = 0; t < 4; ++t) s[t] = c[t] ? S[t][j] * SCALE : NEGV;
      float cm = fmaxf(fmaxf(s[0], s[1]), fmaxf(s[2], s[3]));
      cm = rowred_max(cm);
      float nm = fmaxf(mrow[j], cm);
      float p = __expf(s[0] - nm) + __expf(s[1] - nm) +
                __expf(s[2] - nm) + __expf(s[3] - nm);
      p = rowred_add(p);
      lrow[j] = lrow[j] * __expf(mrow[j] - nm) + p;
      mrow[j] = nm;
    }
  }
  float linv[8];
#pragma unroll
  for (int j = 0; j < 8; ++j) linv[j] = 1.0f / lrow[j];

  // ---- Sweep 2: P = exp(S-m)/l + M*rfro ;  O += P @ V ------------------
  v8f o[8] = {};
  __bf16* myP = sP + wave * (16 * 64);
  for (int k0 = 0; k0 < L_; k0 += 64) {
    __syncthreads();                         // previous chunk fully consumed
#pragma unroll
    for (int q = 0; q < 4; ++q) {
      int e = (tid * 4 + q) * 8;
      cp16_g2l(Kbh + (long long)k0 * D_ + e, sK + e);
      cp16_g2l(kvb + (long long)k0 * D_ + e, sKV + e);
    }
#pragma unroll
    for (int q = 0; q < 4; ++q) {            // V: 128 d-rows x 64 k each
      int idx = tid * 4 + q;                 // 0..1023 16B-granules
      int d = idx >> 3, c8 = (idx & 7) * 8;
      cp16_g2l(Vth + (long long)d * L_ + k0 + c8, sV + d * 64 + c8);
    }
    async_copies_done();
    __syncthreads();

    // sequential 16-col sub-tiles: one S/M accumulator pair live at a time
#pragma unroll
    for (int t = 0; t < 4; ++t) {
      const __bf16* krow = sK  + (t * 16 + lr) * D_;
      const __bf16* mro  = sKV + (t * 16 + lr) * D_;
      v8f St = {}, Mt = {};
#pragma unroll
      for (int i = 0; i < 4; ++i) {
        St = WMMA_BF16(aq[i],  load_b32frag(krow + i * 32 + half * 16), St);
        Mt = WMMA_BF16(aqm[i], load_b32frag(mro  + i * 32 + half * 16), Mt);
      }
      const bool ct = (k0 + t * 16 + lr) < vl;
#pragma unroll
      for (int j = 0; j < 8; ++j) {
        float s = ct ? St[j] * SCALE : NEGV;
        float a = __expf(s - mrow[j]) * linv[j] + Mt[j] * rfro;
        myP[(half * 8 + j) * 64 + t * 16 + lr] = (__bf16)a;
      }
    }
    // wave-private LDS round-trip (C-layout -> A-layout); DS in-order per wave
    asm volatile("s_wait_dscnt 0" ::: "memory");
    Frag f0, f1;
    f0.u[0] = *(const uint4*)(myP + lr * 64 + half * 16);
    f0.u[1] = *(const uint4*)(myP + lr * 64 + 8 + half * 16);
    f1.u[0] = *(const uint4*)(myP + lr * 64 + 32 + half * 16);
    f1.u[1] = *(const uint4*)(myP + lr * 64 + 40 + half * 16);
    v16bf ap0 = f0.v, ap1 = f1.v;
#pragma unroll
    for (int n = 0; n < 8; ++n) {
      const __bf16* vbase = sV + (n * 16 + lr) * 64;
      o[n] = WMMA_BF16(ap0, load_b32frag(vbase + half * 16), o[n]);
      o[n] = WMMA_BF16(ap1, load_b32frag(vbase + 32 + half * 16), o[n]);
    }
  }

  // ---- Write 16x128 fp32 output (C-layout: row = half*8+j, col = n*16+lr)
#pragma unroll
  for (int n = 0; n < 8; ++n)
#pragma unroll
    for (int j = 0; j < 8; ++j)
      out[((long long)bh * L_ + qbase + half * 8 + j) * D_ + n * 16 + lr] = o[n][j];
}

// ---------------------------------------------------------------------------
extern "C" void kernel_launch(void* const* d_in, const int* in_sizes, int n_in,
                              void* d_out, int out_size, void* d_ws, size_t ws_size,
                              hipStream_t stream) {
  const float* Q     = (const float*)d_in[0];
  const float* K     = (const float*)d_in[1];
  const float* V     = (const float*)d_in[2];
  const float* qm    = (const float*)d_in[3];
  const float* kvm   = (const float*)d_in[4];
  const int*   valid = (const int*)d_in[5];
  float* out = (float*)d_out;

  const size_t NQ = (size_t)BH_ * L_ * D_;   // 8,388,608
  const size_t NM = (size_t)B_ * L_ * D_;    //   524,288

  __bf16* Qb   = (__bf16*)d_ws;
  __bf16* Kb   = Qb + NQ;
  __bf16* Vt   = Kb + NQ;
  __bf16* qmb  = Vt + NQ;
  __bf16* kvmb = qmb + NM;
  float*  partial = (float*)(kvmb + NM);     // 32768 floats
  float*  fro     = partial + B_ * 128 * 128;

  convert_kernel<<<2048, 256, 0, stream>>>(Q, K, V, qm, kvm, Qb, Kb, Vt, qmb, kvmb);
  fro_partial_kernel<<<(B_ * 128 * 128) / 8, 256, 0, stream>>>(qmb, kvmb, partial);
  fro_reduce_kernel<<<B_, 256, 0, stream>>>(partial, fro);
  attn_kernel<<<BH_ * 16, 256, 0, stream>>>(Qb, Kb, Vt, qmb, kvmb, valid, fro, out);
}
